// intraview_feature_exchange_62543313764645
// MI455X (gfx1250) — matrix operand order repaired and checked by reading, not compiled, not run on hardware
//
#include <hip/hip_runtime.h>
#include <cstdint>
#include <cstddef>

// ---------------- problem constants ----------------
#define C_DIM   192
#define HEADS   6
#define HEAD_D  32
#define NTOK    99            // 1 global + 98 window tokens
#define NPAD    112           // 7 * 16
#define NWIN    256           // windows per batch: 4*8*8
#define BATCH   2
#define BW      (BATCH*NWIN)  // 512
#define MROWS   (BW*NPAD)     // 57344 (qkv / proj GEMM rows)
#define XROWS   (BATCH*8*56*56) // 50176 spatial tokens
#define ZROWS   (XROWS + BW)    // 50688 MLP rows
#define HIDDEN  768
#define SCALE_Q 0.17677669529663687f   // 32^-0.5
#define NEG_BIG (-3.0e38f)

typedef __attribute__((ext_vector_type(16))) __bf16       bf16x16;
typedef __attribute__((ext_vector_type(8)))  float        f32x8;
typedef __attribute__((ext_vector_type(4)))  unsigned int u32x4;

union Frag { bf16x16 v; u32x4 q[2]; };

__device__ __forceinline__ f32x8 zero8() {
  f32x8 z;
#pragma unroll
  for (int i = 0; i < 8; ++i) z[i] = 0.0f;
  return z;
}

// A-fragment (16x32 bf16, row major, lda in elements)
// lane<16: K = kbase+[0..7] then kbase+16+[0..7]; lane>=16: +8
__device__ __forceinline__ bf16x16 load_frag_a(const __bf16* base, int ld, int row,
                                               int kbase, int lane) {
  int half = (lane >> 4) & 1;
  const __bf16* p = base + (size_t)row * ld + kbase + half * 8;
  Frag f;
  f.q[0] = *(const u32x4*)(p);
  f.q[1] = *(const u32x4*)(p + 16);
  return f.v;
}

// B-fragment (32x16 bf16) read from B^T stored row-major [N, K]:
// lane holds column n, K = kbase+8*half+[0..7] then kbase+16+8*half+[0..7].
__device__ __forceinline__ bf16x16 load_frag_b(const __bf16* base, int ld, int col,
                                               int kbase, int lane) {
  int half = (lane >> 4) & 1;
  const __bf16* p = base + (size_t)col * ld + kbase + half * 8;
  Frag f;
  f.q[0] = *(const u32x4*)(p);
  f.q[1] = *(const u32x4*)(p + 16);
  return f.v;
}

__device__ __forceinline__ f32x8 wmma_bf16(bf16x16 a, bf16x16 b, f32x8 c) {
  return __builtin_amdgcn_wmma_f32_16x16x32_bf16(false, a, false, b, (short)0, c,
                                                 false, false);
}

// One wave computes a 16(M) x 64(N) tile: acc[4] of 16x16.
__device__ __forceinline__ void wave_gemm_16x64(const __bf16* A, int lda,
                                                const __bf16* Bt, int ldb,
                                                int rowBase, int colBase, int Ktot,
                                                int lane, f32x8 acc[4]) {
  int arow = rowBase + (lane & 15);
  for (int k = 0; k < Ktot; k += 32) {
    if (k + 32 < Ktot)
      __builtin_prefetch(A + (size_t)arow * lda + k + 32, 0, 0);
    bf16x16 a = load_frag_a(A, lda, arow, k, lane);
#pragma unroll
    for (int t = 0; t < 4; ++t) {
      bf16x16 b = load_frag_b(Bt, ldb, colBase + t * 16 + (lane & 15), k, lane);
      acc[t] = wmma_bf16(a, b, acc[t]);
    }
  }
}

// ---------------- weight transpose + bf16 convert: in [K,N] f32 -> out [N,K] bf16
__global__ void transpose_bf16_kernel(const float* __restrict__ in,
                                      __bf16* __restrict__ out, int K, int N) {
  int i = blockIdx.x * blockDim.x + threadIdx.x;
  if (i >= K * N) return;
  int k = i / N, n = i % N;
  out[(size_t)n * K + k] = (__bf16)in[i];
}

// ---------------- per-row layernorm (one wave = one 192-ch row) ----------------
__device__ __forceinline__ void ln_row_bf16(const float* __restrict__ src,
                                            const float* __restrict__ w,
                                            const float* __restrict__ b,
                                            __bf16* __restrict__ dst, int lane) {
  float v[6];
#pragma unroll
  for (int i = 0; i < 6; ++i) v[i] = src[lane * 6 + i];
  float s = v[0] + v[1] + v[2] + v[3] + v[4] + v[5];
#pragma unroll
  for (int m = 16; m >= 1; m >>= 1) s += __shfl_xor(s, m, 32);
  float mean = s * (1.0f / 192.0f);
  float q = 0.0f;
#pragma unroll
  for (int i = 0; i < 6; ++i) { float d = v[i] - mean; q += d * d; }
#pragma unroll
  for (int m = 16; m >= 1; m >>= 1) q += __shfl_xor(q, m, 32);
  float rstd = rsqrtf(q * (1.0f / 192.0f) + 1e-5f);
#pragma unroll
  for (int i = 0; i < 6; ++i) {
    int c = lane * 6 + i;
    dst[c] = (__bf16)((v[i] - mean) * rstd * w[c] + b[c]);
  }
}

// ---------------- kernel A: LN + shifted-window gather + global token -> Xw bf16
__global__ void ln_gather_kernel(const float* __restrict__ x,
                                 const float* __restrict__ gt,
                                 const float* __restrict__ n1w,
                                 const float* __restrict__ n1b,
                                 __bf16* __restrict__ Xw) {
  int wave = (blockIdx.x * blockDim.x + threadIdx.x) >> 5;
  int lane = threadIdx.x & 31;
  if (wave >= MROWS) return;
  int b_ = wave / NPAD, t = wave % NPAD;
  __bf16* dst = Xw + (size_t)wave * C_DIM;
  if (t == 0) {  // global token (not layer-normed)
#pragma unroll
    for (int i = 0; i < 6; ++i)
      dst[lane * 6 + i] = (__bf16)gt[(size_t)b_ * C_DIM + lane * 6 + i];
    return;
  }
  if (t > 98) {  // pad row
#pragma unroll
    for (int i = 0; i < 6; ++i) dst[lane * 6 + i] = (__bf16)0.0f;
    return;
  }
  int tl = t - 1;
  int td = tl / 49, th = (tl / 7) % 7, tw = tl % 7;
  int b = b_ >> 8, widx = b_ & 255;
  int dw = widx >> 6, hw = (widx >> 3) & 7, ww = widx & 7;
  int sd = (dw * 2 + td + 1) & 7;          // undo roll(-1) on D (size 8)
  int sh = (hw * 7 + th + 3) % 56;         // undo roll(-3) on H
  int sw = (ww * 7 + tw + 3) % 56;         // undo roll(-3) on W
  const float* src = x + (size_t)(((b * 8 + sd) * 56 + sh) * 56 + sw) * C_DIM;
  ln_row_bf16(src, n1w, n1b, dst, lane);
}

// ---------------- kernel B: QKV GEMM, split/scale/relayout epilog ----------------
__global__ void gemm_qkv_kernel(const __bf16* __restrict__ Xw,
                                const __bf16* __restrict__ Wt,
                                const float* __restrict__ bias,
                                __bf16* __restrict__ Q, __bf16* __restrict__ K,
                                __bf16* __restrict__ V) {
  int wv = threadIdx.x >> 5, lane = threadIdx.x & 31;
  int rowBase = blockIdx.x * 128 + wv * 16;
  int colBase = blockIdx.y * 64;
  f32x8 acc[4];
#pragma unroll
  for (int t = 0; t < 4; ++t) acc[t] = zero8();
  wave_gemm_16x64(Xw, C_DIM, Wt, C_DIM, rowBase, colBase, C_DIM, lane, acc);
  int half = lane >> 4, cl = lane & 15;
#pragma unroll
  for (int t = 0; t < 4; ++t) {
    int col = colBase + t * 16 + cl;
    int part = col / C_DIM, rem = col % C_DIM;
    int hd = rem >> 5, dim = rem & 31;
    __bf16* dst = (part == 0) ? Q : (part == 1) ? K : V;
#pragma unroll
    for (int r = 0; r < 8; ++r) {
      int row = rowBase + r + 8 * half;
      float v = acc[t][r] + bias[col];
      if (part == 0) v *= SCALE_Q;
      int b_ = row / NPAD, tok = row % NPAD;
      dst[(((size_t)b_ * HEADS + hd) * NPAD + tok) * HEAD_D + dim] = (__bf16)v;
    }
  }
}

// ---------------- kernel C: windowed attention (one block = window*head) --------
__global__ void attn_kernel(const __bf16* __restrict__ Q,
                            const __bf16* __restrict__ K,
                            const __bf16* __restrict__ V,
                            const float* __restrict__ rpb,
                            const float* __restrict__ amask,
                            __bf16* __restrict__ Xattn) {
  __shared__ __bf16 Pld[NPAD][128];   // softmax probs, cols padded to 128
  __shared__ __bf16 Vt[HEAD_D][128];  // V transposed, tokens padded to 128

  int wv = threadIdx.x >> 5, lane = threadIdx.x & 31;
  int bh = blockIdx.x;
  int b_ = bh / HEADS, h = bh % HEADS;
  int wmod = b_ & 255;
  const __bf16* Qb = Q + (size_t)bh * NPAD * HEAD_D;
  const __bf16* Kb = K + (size_t)bh * NPAD * HEAD_D;
  const __bf16* Vb = V + (size_t)bh * NPAD * HEAD_D;

  // stage V^T into LDS (zero-pad tokens 112..127)
  for (int i = threadIdx.x; i < HEAD_D * 128; i += 256) {
    int dim = i >> 7, tok = i & 127;
    Vt[dim][tok] = (tok < NPAD) ? Vb[(size_t)tok * HEAD_D + dim] : (__bf16)0.0f;
  }

  if (wv < 7) {  // S = Q K^T : wave wv owns the 16-row tile wv
    f32x8 acc[7];
#pragma unroll
    for (int j = 0; j < 7; ++j) acc[j] = zero8();
    bf16x16 qf = load_frag_a(Qb, HEAD_D, wv * 16 + (lane & 15), 0, lane);
#pragma unroll
    for (int j = 0; j < 7; ++j) {
      bf16x16 kf = load_frag_b(Kb, HEAD_D, j * 16 + (lane & 15), 0, lane);
      acc[j] = wmma_bf16(qf, kf, acc[j]);
    }
    int half = lane >> 4, cl = lane & 15;
    // rel-pos bias + window mask + column padding mask
#pragma unroll
    for (int j = 0; j < 7; ++j) {
      int jc = j * 16 + cl;
#pragma unroll
      for (int r = 0; r < 8; ++r) {
        int i = wv * 16 + r + 8 * half;
        float s = acc[j][r];
        if (jc >= NTOK) {
          s = NEG_BIG;
        } else if (i >= 1 && i <= 98 && jc >= 1) {
          int ii = i - 1, jj = jc - 1;
          int di = ii / 49 - jj / 49 + 1;
          int dh = (ii / 7) % 7 - (jj / 7) % 7 + 6;
          int dv = ii % 7 - jj % 7 + 6;
          int ridx = (di * 13 + dh) * 13 + dv;
          s += rpb[ridx * HEADS + h] + amask[((size_t)wmod * 98 + ii) * 98 + jj];
        }
        acc[j][r] = s;
      }
    }
    // row softmax: each logical row lives in one 16-lane half
#pragma unroll
    for (int r = 0; r < 8; ++r) {
      float m = NEG_BIG;
#pragma unroll
      for (int j = 0; j < 7; ++j) m = fmaxf(m, acc[j][r]);
#pragma unroll
      for (int msk = 8; msk >= 1; msk >>= 1) m = fmaxf(m, __shfl_xor(m, msk, 32));
      float sum = 0.0f;
#pragma unroll
      for (int j = 0; j < 7; ++j) {
        float e = __expf(acc[j][r] - m);
        acc[j][r] = e;
        sum += e;
      }
#pragma unroll
      for (int msk = 8; msk >= 1; msk >>= 1) sum += __shfl_xor(sum, msk, 32);
      float inv = 1.0f / sum;
      int i = wv * 16 + r + 8 * half;
#pragma unroll
      for (int j = 0; j < 7; ++j) Pld[i][j * 16 + cl] = (__bf16)(acc[j][r] * inv);
      // zero only LDS padding column 112 + i % 16 of row i (cols 99..111 are exp(-inf)=0)
      if (cl == 0) Pld[i][112 + half * 8 + r] = (__bf16)0.0f;
    }
    // via half/r each of the 16 rows zeroes a different one of cols 112..127; the rest stay unwritten
  }
  __syncthreads();

  // O = P V : 14 tiles (7 M x 2 N) over 8 waves
  for (int t = wv; t < 14; t += 8) {
    int m = t % 7, n = t / 7;
    f32x8 acc = zero8();
#pragma unroll
    for (int ks = 0; ks < 4; ++ks) {
      bf16x16 a = load_frag_a(&Pld[0][0], 128, m * 16 + (lane & 15), ks * 32, lane);
      bf16x16 b = load_frag_b(&Vt[0][0], 128, n * 16 + (lane & 15), ks * 32, lane);
      acc = wmma_bf16(a, b, acc);
    }
    int half = lane >> 4, cl = lane & 15;
#pragma unroll
    for (int r = 0; r < 8; ++r) {
      int tok = m * 16 + r + 8 * half;
      if (tok <= 98)
        Xattn[((size_t)b_ * NPAD + tok) * C_DIM + h * HEAD_D + n * 16 + cl] =
            (__bf16)acc[r];
    }
  }
}

// ---------------- kernel D: proj GEMM + window-reverse scatter + residual -------
__global__ void gemm_proj_kernel(const __bf16* __restrict__ Xattn,
                                 const __bf16* __restrict__ Wt,
                                 const float* __restrict__ bias,
                                 const float* __restrict__ x_in,
                                 float* __restrict__ y, float* __restrict__ gtp) {
  int wv = threadIdx.x >> 5, lane = threadIdx.x & 31;
  int rowBase = blockIdx.x * 128 + wv * 16;
  int colBase = blockIdx.y * 64;
  f32x8 acc[4];
#pragma unroll
  for (int t = 0; t < 4; ++t) acc[t] = zero8();
  wave_gemm_16x64(Xattn, C_DIM, Wt, C_DIM, rowBase, colBase, C_DIM, lane, acc);
  int half = lane >> 4, cl = lane & 15;
#pragma unroll
  for (int t = 0; t < 4; ++t) {
    int col = colBase + t * 16 + cl;
#pragma unroll
    for (int r = 0; r < 8; ++r) {
      int row = rowBase + r + 8 * half;
      float v = acc[t][r] + bias[col];
      int b_ = row / NPAD, tok = row % NPAD;
      if (tok == 0) {
        gtp[(size_t)b_ * C_DIM + col] = v;
      } else if (tok <= 98) {
        int tl = tok - 1;
        int td = tl / 49, th = (tl / 7) % 7, tw = tl % 7;
        int b = b_ >> 8, widx = b_ & 255;
        int dw = widx >> 6, hw = (widx >> 3) & 7, ww = widx & 7;
        int sd = (dw * 2 + td + 1) & 7;
        int sh = (hw * 7 + th + 3) % 56;
        int sw = (ww * 7 + tw + 3) % 56;
        size_t idx = (size_t)(((b * 8 + sd) * 56 + sh) * 56 + sw) * C_DIM + col;
        y[idx] = x_in[idx] + v;
      }
    }
  }
}

// ---------------- kernel E1: LN for MLP input (x rows then gt rows) -------------
__global__ void ln_mlp_kernel(const float* __restrict__ y,
                              const float* __restrict__ gtp,
                              const float* __restrict__ n2w,
                              const float* __restrict__ n2b,
                              __bf16* __restrict__ Zn) {
  int wave = (blockIdx.x * blockDim.x + threadIdx.x) >> 5;
  int lane = threadIdx.x & 31;
  if (wave >= ZROWS) return;
  const float* src = (wave < XROWS) ? (y + (size_t)wave * C_DIM)
                                    : (gtp + (size_t)(wave - XROWS) * C_DIM);
  ln_row_bf16(src, n2w, n2b, Zn + (size_t)wave * C_DIM, lane);
}

// ---------------- kernel E2: fc1 GEMM + exact GELU ----------------
__global__ void gemm_fc1_kernel(const __bf16* __restrict__ Zn,
                                const __bf16* __restrict__ Wt,
                                const float* __restrict__ bias,
                                __bf16* __restrict__ Hb) {
  int wv = threadIdx.x >> 5, lane = threadIdx.x & 31;
  int rowBase = blockIdx.x * 128 + wv * 16;
  int colBase = blockIdx.y * 64;
  f32x8 acc[4];
#pragma unroll
  for (int t = 0; t < 4; ++t) acc[t] = zero8();
  wave_gemm_16x64(Zn, C_DIM, Wt, C_DIM, rowBase, colBase, C_DIM, lane, acc);
  int half = lane >> 4, cl = lane & 15;
#pragma unroll
  for (int t = 0; t < 4; ++t) {
    int col = colBase + t * 16 + cl;
#pragma unroll
    for (int r = 0; r < 8; ++r) {
      int row = rowBase + r + 8 * half;
      float v = acc[t][r] + bias[col];
      v = 0.5f * v * (1.0f + erff(v * 0.7071067811865475f));
      Hb[(size_t)row * HIDDEN + col] = (__bf16)v;
    }
  }
}

// ---------------- kernel E3: fc2 GEMM + residual -> d_out ----------------
__global__ void gemm_fc2_kernel(const __bf16* __restrict__ Hb,
                                const __bf16* __restrict__ Wt,
                                const float* __restrict__ bias,
                                const float* __restrict__ y,
                                const float* __restrict__ gtp,
                                float* __restrict__ out) {
  int wv = threadIdx.x >> 5, lane = threadIdx.x & 31;
  int rowBase = blockIdx.x * 128 + wv * 16;
  int colBase = blockIdx.y * 64;
  f32x8 acc[4];
#pragma unroll
  for (int t = 0; t < 4; ++t) acc[t] = zero8();
  wave_gemm_16x64(Hb, HIDDEN, Wt, HIDDEN, rowBase, colBase, HIDDEN, lane, acc);
  int half = lane >> 4, cl = lane & 15;
#pragma unroll
  for (int t = 0; t < 4; ++t) {
    int col = colBase + t * 16 + cl;
#pragma unroll
    for (int r = 0; r < 8; ++r) {
      int row = rowBase + r + 8 * half;
      size_t idx = (size_t)row * C_DIM + col;
      float res = (row < XROWS) ? y[idx] : gtp[idx - (size_t)XROWS * C_DIM];
      out[idx] = res + acc[t][r] + bias[col];
    }
  }
}

// ---------------- workspace layout (bytes); total requirement ~146 MB ----------
#define MB (1024ull * 1024ull)
static const size_t OFF_WQ  = 0 * MB;   // qkv_w^T  bf16 [576,192]   (0.22 MB)
static const size_t OFF_WP  = 1 * MB;   // proj_w^T bf16 [192,192]   (0.07 MB)
static const size_t OFF_WF1 = 2 * MB;   // fc1_w^T  bf16 [768,192]   (0.29 MB)
static const size_t OFF_WF2 = 3 * MB;   // fc2_w^T  bf16 [192,768]   (0.29 MB)
static const size_t OFF_XW  = 4 * MB;   // Xw bf16 [57344,192] 22MB; reused: Xattn, Zn
static const size_t OFF_Q   = 28 * MB;  // Q bf16 22MB; later reused (with K) as y f32
static const size_t OFF_K   = 52 * MB;  // K bf16 22MB
static const size_t OFF_V   = 76 * MB;  // V bf16 22MB (dead after attention)
static const size_t OFF_Y   = OFF_Q;    // y f32 [50176,192] 38.5MB (over Q+K)
static const size_t OFF_GTP = 67 * MB;  // gt proj f32 [512,192] 0.38MB
static const size_t OFF_HB  = 68 * MB;  // Hb bf16 [50688,768] 77.9MB -> end ~146MB

extern "C" void kernel_launch(void* const* d_in, const int* in_sizes, int n_in,
                              void* d_out, int out_size, void* d_ws, size_t ws_size,
                              hipStream_t stream) {
  const float* x    = (const float*)d_in[0];
  const float* gt   = (const float*)d_in[1];
  // d_in[2] = position : unused by the reference
  const float* amsk = (const float*)d_in[3];
  const float* n1w  = (const float*)d_in[4];
  const float* n1b  = (const float*)d_in[5];
  const float* qkvw = (const float*)d_in[6];
  const float* qkvb = (const float*)d_in[7];
  const float* prjw = (const float*)d_in[8];
  const float* prjb = (const float*)d_in[9];
  const float* rpb  = (const float*)d_in[10];
  const float* n2w  = (const float*)d_in[11];
  const float* n2b  = (const float*)d_in[12];
  const float* f1w  = (const float*)d_in[13];
  const float* f1b  = (const float*)d_in[14];
  const float* f2w  = (const float*)d_in[15];
  const float* f2b  = (const float*)d_in[16];

  char* ws = (char*)d_ws;
  __bf16* wQ  = (__bf16*)(ws + OFF_WQ);
  __bf16* wP  = (__bf16*)(ws + OFF_WP);
  __bf16* wF1 = (__bf16*)(ws + OFF_WF1);
  __bf16* wF2 = (__bf16*)(ws + OFF_WF2);
  __bf16* Xw  = (__bf16*)(ws + OFF_XW);
  __bf16* Qm  = (__bf16*)(ws + OFF_Q);
  __bf16* Km  = (__bf16*)(ws + OFF_K);
  __bf16* Vm  = (__bf16*)(ws + OFF_V);
  __bf16* Xat = (__bf16*)(ws + OFF_XW);   // reuse
  float*  y   = (float*)(ws + OFF_Y);     // reuse (Q+K)
  float*  gtp = (float*)(ws + OFF_GTP);
  __bf16* Zn  = (__bf16*)(ws + OFF_XW);   // reuse
  __bf16* Hb  = (__bf16*)(ws + OFF_HB);
  float*  out = (float*)d_out;

  // weight pre-transpose to [N,K] bf16
  transpose_bf16_kernel<<<(192 * 576 + 255) / 256, 256, 0, stream>>>(qkvw, wQ, 192, 576);
  transpose_bf16_kernel<<<(192 * 192 + 255) / 256, 256, 0, stream>>>(prjw, wP, 192, 192);
  transpose_bf16_kernel<<<(192 * 768 + 255) / 256, 256, 0, stream>>>(f1w, wF1, 192, 768);
  transpose_bf16_kernel<<<(768 * 192 + 255) / 256, 256, 0, stream>>>(f2w, wF2, 768, 192);

  // LN + shifted-window gather (+global token) -> Xw
  ln_gather_kernel<<<MROWS / 8, 256, 0, stream>>>(x, gt, n1w, n1b, Xw);

  // QKV GEMM: [57344,192] x [192,576]
  gemm_qkv_kernel<<<dim3(MROWS / 128, 576 / 64), 256, 0, stream>>>(Xw, wQ, qkvb,
                                                                   Qm, Km, Vm);

  // attention per (window, head)
  attn_kernel<<<BW * HEADS, 256, 0, stream>>>(Qm, Km, Vm, rpb, amsk, Xat);

  // proj GEMM + window-reverse scatter + first residual
  gemm_proj_kernel<<<dim3(MROWS / 128, 192 / 64), 256, 0, stream>>>(Xat, wP, prjb,
                                                                    x, y, gtp);

  // MLP (x rows + gt rows together)
  ln_mlp_kernel<<<(ZROWS + 7) / 8, 256, 0, stream>>>(y, gtp, n2w, n2b, Zn);
  gemm_fc1_kernel<<<dim3(ZROWS / 128, HIDDEN / 64), 256, 0, stream>>>(Zn, wF1, f1b, Hb);
  gemm_fc2_kernel<<<dim3(ZROWS / 128, 192 / 64), 256, 0, stream>>>(Hb, wF2, f2b,
                                                                   y, gtp, out);
}
